// Head_31645319037145
// MI455X (gfx1250) — matrix operand order repaired and checked by reading, not compiled, not run on hardware
//
#include <hip/hip_runtime.h>
#include <hip/hip_bf16.h>

// Flash-attention-style fused head for gfx1250 (MI455X), wave32 WMMA.
//   B=16, T=2048, C=64.  out = softmax(q k^T * sqrt(C)) v, q/k/v = x W + b.
// Compute-bound (~18 GFLOP vs ~30MB traffic) -> f16 WMMA w/ f32 accum,
// online softmax, S^T = K Q^T trick so P (D-frag) remaps to A-frag in-lane.
// All fragment loads are 16B-aligned v8h runs -> global_load_b128; K and V
// tile loads are hoisted ahead of the exp stretch so VMEM latency overlaps
// the transcendental pipeline.

typedef __attribute__((ext_vector_type(8)))  _Float16 v8h;
typedef __attribute__((ext_vector_type(16))) _Float16 v16h;
typedef __attribute__((ext_vector_type(8)))  float    v8f;
typedef __attribute__((ext_vector_type(4)))  float    v4f;

#define NB 16
#define TT 2048
#define CC 64

static __device__ __forceinline__ v8f wmma16(const v16h a, const v16h b, const v8f c) {
    // D = A(16x32 f16) * B(32x16 f16) + C(16x16 f32)
    return __builtin_amdgcn_wmma_f32_16x16x32_f16(
        /*neg_a=*/false, a, /*neg_b=*/false, b,
        /*c_mod=*/(short)0, c, /*reuse_a=*/false, /*reuse_b=*/false);
}
static __device__ __forceinline__ v8h ld8(const _Float16* p) {
    return *(const v8h*)p;                 // 16B-aligned by construction
}
static __device__ __forceinline__ v16h cat(v8h a, v8h b) {
    return __builtin_shufflevector(a, b, 0,1,2,3,4,5,6,7,8,9,10,11,12,13,14,15);
}

// ---------------------------------------------------------------------------
// Kernel 0: convert weights to f16, transposed [d][c] so projection B-frags
// are contiguous 8-half runs.  Wt = 3 x 64 x 64 halves.
// ---------------------------------------------------------------------------
__global__ __launch_bounds__(256) void wprep_kernel(
    const float* __restrict__ Wq, const float* __restrict__ Wk,
    const float* __restrict__ Wv, _Float16* __restrict__ Wt)
{
    int idx = blockIdx.x * 256 + threadIdx.x;          // 0 .. 3*4096-1
    if (idx >= 3 * CC * CC) return;
    int p = idx >> 12;
    int r = idx & 4095;
    int d = r >> 6, c = r & 63;
    const float* W = (p == 0) ? Wq : ((p == 1) ? Wk : Wv);
    Wt[idx] = (_Float16)W[c * CC + d];                 // Wt[p][d][c] = W[c][d]
}

// ---------------------------------------------------------------------------
// Kernel 1: QKV projection.  One wave handles 16 rows of flattened x [B*T, 64].
// Q,K stored row-major f16 [B*T][64]; V stored transposed f16 [B][64][T].
// ---------------------------------------------------------------------------
__global__ __launch_bounds__(128) void qkv_proj_kernel(
    const float* __restrict__ x,
    const float* __restrict__ bk, const float* __restrict__ bq,
    const float* __restrict__ bv, const _Float16* __restrict__ Wt,
    _Float16* __restrict__ Qh, _Float16* __restrict__ Kh, _Float16* __restrict__ Vt)
{
    const int lane = threadIdx.x & 31;
    const int wave = threadIdx.x >> 5;
    const int tile = blockIdx.x * 4 + wave;       // 16-row tile index over B*T/16
    const int row0 = tile * 16;
    const int m    = lane & 15;
    const int half = lane >> 4;
    const int batch = row0 >> 11;                 // row0 / T
    const int t0    = row0 & (TT - 1);

    // A-fragments of x tile: elem h -> c = (h>>3)*16 + (h&7) + 8*half + 32*ch
    // Two 8-float runs per chunk, loaded as v4f pairs (b128).
    v16h XA[2];
    const float* rp = x + (size_t)(row0 + m) * CC;
#pragma unroll
    for (int ch = 0; ch < 2; ++ch) {
        int base = 8 * half + 32 * ch;
        v4f a0 = *(const v4f*)(rp + base);
        v4f a1 = *(const v4f*)(rp + base + 4);
        v4f b0 = *(const v4f*)(rp + base + 16);
        v4f b1 = *(const v4f*)(rp + base + 20);
#pragma unroll
        for (int j = 0; j < 4; ++j) {
            XA[ch][j]      = (_Float16)a0[j];
            XA[ch][4 + j]  = (_Float16)a1[j];
            XA[ch][8 + j]  = (_Float16)b0[j];
            XA[ch][12 + j] = (_Float16)b1[j];
        }
    }

    const float* biases[3] = {bq, bk, bv};

#pragma unroll
    for (int p = 0; p < 3; ++p) {
        const _Float16* Wtp = Wt + p * CC * CC;   // [d][c] f16
        const float*    bias = biases[p];
#pragma unroll
        for (int nt = 0; nt < 4; ++nt) {          // output column tile (N=16)
            const int col = m + 16 * nt;
            // B layout: N = lane&15 ; elem h -> kk(c) = (h>>3)*16 + (h&7) + 8*half + 32*ch
            v16h Bf[2];
#pragma unroll
            for (int ch = 0; ch < 2; ++ch) {
                const _Float16* wp = Wtp + (size_t)col * CC + 8 * half + 32 * ch;
                Bf[ch] = cat(ld8(wp), ld8(wp + 16));
            }
            v8f acc = {};
            acc = wmma16(XA[0], Bf[0], acc);
            acc = wmma16(XA[1], Bf[1], acc);
            const float bval = bias[col];
            // D layout: row = i + 8*half, col = lane&15 + 16*nt
#pragma unroll
            for (int i = 0; i < 8; ++i) {
                float val = acc[i] + bval;
                int r = i + 8 * half;
                if (p == 0)      Qh[(size_t)(row0 + r) * CC + col] = (_Float16)val;
                else if (p == 1) Kh[(size_t)(row0 + r) * CC + col] = (_Float16)val;
                else             Vt[((size_t)batch * CC + col) * TT + (t0 + r)] = (_Float16)val;
            }
        }
    }
}

// ---------------------------------------------------------------------------
// Kernel 2: fused attention.  One wave owns TWO 16-query tiles (32 queries);
// K/V fragment loads are shared by both tiles.  Iterates over 64-key tiles
// with online softmax.  S^T = K*Q^T so P D-frags remap to A-frags in-lane.
// ---------------------------------------------------------------------------
__global__ __launch_bounds__(128) void attn_kernel(
    const _Float16* __restrict__ Qh, const _Float16* __restrict__ Kh,
    const _Float16* __restrict__ Vt, float* __restrict__ out)
{
    const int lane = threadIdx.x & 31;
    const int wave = threadIdx.x >> 5;
    const int tile = blockIdx.x * 4 + wave;       // over B*T/32 = 1024 tiles
    const int batch = tile >> 6;                  // 64 tiles of 32 queries per batch
    const int q0    = (tile & 63) * 32;
    const int m     = lane & 15;
    const int half  = lane >> 4;

    const _Float16* Qb = Qh + (size_t)batch * TT * CC;
    const _Float16* Kb = Kh + (size_t)batch * TT * CC;
    const _Float16* Vb = Vt + (size_t)batch * CC * TT;

    // Q^T B-frags: elem h -> c = (h>>3)*16 + (h&7) + 8*half + 32*ch ; n(query) = lane&15
    v16h QB[2][2];
#pragma unroll
    for (int qt = 0; qt < 2; ++qt)
#pragma unroll
        for (int ch = 0; ch < 2; ++ch) {
            const _Float16* qp = Qb + (size_t)(q0 + 16 * qt + m) * CC + 8 * half + 32 * ch;
            QB[qt][ch] = cat(ld8(qp), ld8(qp + 16));
        }

    v8f acc[2][4];
#pragma unroll
    for (int qt = 0; qt < 2; ++qt)
#pragma unroll
        for (int cn = 0; cn < 4; ++cn) acc[qt][cn] = (v8f){};
    float mrun[2] = {-1e30f, -1e30f};
    float lrun[2] = {0.0f, 0.0f};

    for (int kb = 0; kb < TT; kb += 64) {
        if (kb + 64 < TT) {
            __builtin_prefetch(&Kb[(size_t)(kb + 64 + m) * CC], 0, 1);
            __builtin_prefetch(&Vb[(size_t)m * TT + kb + 64], 0, 1);
        }
        // ---- load whole 64-key K tile as A-frags (b128 clauses) ----
        // A layout: M = lane&15 ; elem h -> c = (h>>3)*16 + (h&7) + 8*half + 32*ch
        v16h KA[4][2];
#pragma unroll
        for (int nt = 0; nt < 4; ++nt)
#pragma unroll
            for (int ch = 0; ch < 2; ++ch) {
                const _Float16* kp = Kb + (size_t)(kb + nt * 16 + m) * CC + 8 * half + 32 * ch;
                KA[nt][ch] = cat(ld8(kp), ld8(kp + 16));
            }
        // ---- hoist V tile B-frags: independent of softmax, so their latency
        //      is covered by the exp/transcendental stretch below ----
        // B layout: N(c) = lane&15 + 16*cn ; elem h -> key = (h>>3)*16 + (h&7) + 8*half + 32*ch
        v16h VB[4][2];
#pragma unroll
        for (int cn = 0; cn < 4; ++cn)
#pragma unroll
            for (int ch = 0; ch < 2; ++ch) {
                const _Float16* vp = Vb + (size_t)(m + 16 * cn) * TT + kb + 32 * ch + 8 * half;
                VB[cn][ch] = cat(ld8(vp), ld8(vp + 16));
            }
        // ---- S^T = K Q^T, times sqrt(C)=8 (reference MULTIPLIES) ----
        v8f S[2][4];
#pragma unroll
        for (int qt = 0; qt < 2; ++qt)
#pragma unroll
            for (int nt = 0; nt < 4; ++nt) {
                v8f s = {};
                s = wmma16(KA[nt][0], QB[qt][0], s);
                s = wmma16(KA[nt][1], QB[qt][1], s);
#pragma unroll
                for (int i = 0; i < 8; ++i) s[i] *= 8.0f;
                S[qt][nt] = s;
            }
        // ---- online softmax per query tile; query = lane&15 ----
        float aB[2][8];
        v16h PA[2][2];
#pragma unroll
        for (int qt = 0; qt < 2; ++qt) {
            float loc = -1e30f;
#pragma unroll
            for (int nt = 0; nt < 4; ++nt)
#pragma unroll
                for (int i = 0; i < 8; ++i) loc = fmaxf(loc, S[qt][nt][i]);
            loc = fmaxf(loc, __shfl_xor(loc, 16, 32));
            const float mnew  = fmaxf(mrun[qt], loc);
            const float alpha = __expf(mrun[qt] - mnew);
            float lsum = 0.0f;
#pragma unroll
            for (int nt = 0; nt < 4; ++nt)
#pragma unroll
                for (int i = 0; i < 8; ++i) {
                    float e = __expf(S[qt][nt][i] - mnew);
                    S[qt][nt][i] = e;
                    lsum += e;
                }
            lsum += __shfl_xor(lsum, 16, 32);
            lrun[qt] = lrun[qt] * alpha + lsum;
            mrun[qt] = mnew;
            // P D-frags -> A-frags: pure in-lane register remap
#pragma unroll
            for (int h = 0; h < 16; ++h) {
                PA[qt][0][h] = (_Float16)S[qt][(h >> 3)][h & 7];     // keys kb+0..31
                PA[qt][1][h] = (_Float16)S[qt][2 + (h >> 3)][h & 7]; // keys kb+32..63
            }
#pragma unroll
            for (int i = 0; i < 8; ++i) aB[qt][i] = __shfl(alpha, i + 8 * half, 32);
        }
        // ---- P * V accumulate; V frags shared by both query tiles ----
#pragma unroll
        for (int cn = 0; cn < 4; ++cn) {
#pragma unroll
            for (int qt = 0; qt < 2; ++qt) {
                v8f o = acc[qt][cn];
#pragma unroll
                for (int i = 0; i < 8; ++i) o[i] *= aB[qt][i];
                o = wmma16(PA[qt][0], VB[cn][0], o);
                o = wmma16(PA[qt][1], VB[cn][1], o);
                acc[qt][cn] = o;
            }
        }
    }

    // ---- finalize: divide by row sums, write f32 out [B,T,C] ----
    float* ob = out + (size_t)batch * TT * CC;
#pragma unroll
    for (int qt = 0; qt < 2; ++qt) {
        float lB[8];
#pragma unroll
        for (int i = 0; i < 8; ++i) lB[i] = __shfl(lrun[qt], i + 8 * half, 32);
#pragma unroll
        for (int cn = 0; cn < 4; ++cn)
#pragma unroll
            for (int i = 0; i < 8; ++i)
                ob[(size_t)(q0 + 16 * qt + i + 8 * half) * CC + (m + 16 * cn)] =
                    acc[qt][cn][i] / lB[i];
    }
}

// ---------------------------------------------------------------------------
extern "C" void kernel_launch(void* const* d_in, const int* in_sizes, int n_in,
                              void* d_out, int out_size, void* d_ws, size_t ws_size,
                              hipStream_t stream) {
    (void)in_sizes; (void)n_in; (void)out_size; (void)ws_size;
    const float* x  = (const float*)d_in[0];
    const float* Wk = (const float*)d_in[1];
    const float* bk = (const float*)d_in[2];
    const float* Wq = (const float*)d_in[3];
    const float* bq = (const float*)d_in[4];
    const float* Wv = (const float*)d_in[5];
    const float* bv = (const float*)d_in[6];

    // Workspace: f16 Q [B*T][64], K [B*T][64], V^T [B][64][T], Wt 3x64x64
    _Float16* Qh = (_Float16*)d_ws;
    _Float16* Kh = Qh + (size_t)NB * TT * CC;
    _Float16* Vt = Kh + (size_t)NB * TT * CC;
    _Float16* Wt = Vt + (size_t)NB * TT * CC;

    wprep_kernel<<<(3 * CC * CC + 255) / 256, 256, 0, stream>>>(Wq, Wk, Wv, Wt);

    const int ptiles = NB * TT / 16;   // 2048 16-row tiles
    qkv_proj_kernel<<<ptiles / 4, 128, 0, stream>>>(x, bk, bq, bv, Wt, Qh, Kh, Vt);

    const int atiles = NB * TT / 32;   // 1024 32-query tiles
    attn_kernel<<<atiles / 4, 128, 0, stream>>>(Qh, Kh, Vt, (float*)d_out);
}
